// MoELayer_70446053589281
// MI455X (gfx1250) — compile-verified, matrix operand order checked by reading, not run on hardware
//
#include <hip/hip_runtime.h>
#include <hip/hip_bf16.h>

#define HDIM 2048
#define FDIM 8192
#define NEXP 8
#define TOPK 2
#define CAPACITY 512

typedef __attribute__((ext_vector_type(16))) __bf16 v16bf;
typedef __attribute__((ext_vector_type(8)))  float  v8f;

__device__ __forceinline__ unsigned short f32_to_bf16(float f) {
    unsigned int u = __float_as_uint(f);
    unsigned int r = 0x7FFFu + ((u >> 16) & 1u);   // round-to-nearest-even
    u += r;
    return (unsigned short)(u >> 16);
}

__device__ __forceinline__ float gelu_exact(float x) {
    return 0.5f * x * (1.0f + erff(x * 0.70710678118654752f));
}

// CDNA5 async global->LDS copy (16B per lane), tracked by ASYNCcnt.
// Generic pointers into __shared__ carry the LDS byte offset in addr[31:0]
// (ISA 10.2 aperture rules), so truncation yields the DS address.
__device__ __forceinline__ void async_ld_b128(void* lds_dst, const void* gsrc) {
    unsigned l = (unsigned)(unsigned long long)lds_dst;
    asm volatile("global_load_async_to_lds_b128 %0, %1, off"
                 :: "v"(l), "v"(gsrc) : "memory");
}
__device__ __forceinline__ void wait_asynccnt0() {
    asm volatile("s_wait_asynccnt 0" ::: "memory");
}

// ---------------------------------------------------------------------------
// fp32 -> bf16 bulk convert (vectorized: float4 in, uint2 out)
// ---------------------------------------------------------------------------
__global__ __launch_bounds__(256)
void cvt_bf16_kernel(const float* __restrict__ src, unsigned short* __restrict__ dst,
                     long n4) {
    long i = (long)blockIdx.x * blockDim.x + threadIdx.x;
    long stride = (long)gridDim.x * blockDim.x;
    const float4* s4 = reinterpret_cast<const float4*>(src);
    uint2* d2 = reinterpret_cast<uint2*>(dst);
    for (; i < n4; i += stride) {
        float4 f = s4[i];
        uint2 o;
        o.x = (unsigned)f32_to_bf16(f.x) | ((unsigned)f32_to_bf16(f.y) << 16);
        o.y = (unsigned)f32_to_bf16(f.z) | ((unsigned)f32_to_bf16(f.w) << 16);
        d2[i] = o;
    }
}

// ---------------------------------------------------------------------------
// Router: one wave32 per token. 8 logit dot-products over H, wave reduction,
// softmax + top-2 on lane 0.
// ---------------------------------------------------------------------------
__global__ __launch_bounds__(256)
void router_kernel(const float* __restrict__ x, const float* __restrict__ rw,
                   int* __restrict__ topi, float* __restrict__ topp, int N) {
    int wave = threadIdx.x >> 5;
    int lane = threadIdx.x & 31;
    int t = blockIdx.x * 8 + wave;
    if (t >= N) return;
    float acc[NEXP];
#pragma unroll
    for (int e = 0; e < NEXP; e++) acc[e] = 0.0f;
    const float* xp = x + (long)t * HDIM;
    for (int h = lane; h < HDIM; h += 32) {
        float xv = xp[h];
#pragma unroll
        for (int e = 0; e < NEXP; e++) acc[e] += xv * rw[e * HDIM + h];
    }
#pragma unroll
    for (int e = 0; e < NEXP; e++) {
        for (int off = 16; off >= 1; off >>= 1)
            acc[e] += __shfl_xor(acc[e], off, 32);
    }
    if (lane == 0) {
        float m = acc[0];
#pragma unroll
        for (int e = 1; e < NEXP; e++) m = fmaxf(m, acc[e]);
        float p[NEXP], s = 0.0f;
#pragma unroll
        for (int e = 0; e < NEXP; e++) { p[e] = __expf(acc[e] - m); s += p[e]; }
        float inv = 1.0f / s;
        int i0 = 0; float p0 = p[0];
#pragma unroll
        for (int e = 1; e < NEXP; e++) if (p[e] > p0) { p0 = p[e]; i0 = e; }
        int i1 = -1; float p1 = -1.0f;
#pragma unroll
        for (int e = 0; e < NEXP; e++) if (e != i0 && p[e] > p1) { p1 = p[e]; i1 = e; }
        topi[t * 2 + 0] = i0;  topi[t * 2 + 1] = i1;
        topp[t * 2 + 0] = p0 * inv;  topp[t * 2 + 1] = p1 * inv;
    }
}

// ---------------------------------------------------------------------------
// Capacity scan: lane e owns expert e; serial scan over i = k*N + n in the
// reference's slot-major order -> exact deterministic cumsum positions.
// ---------------------------------------------------------------------------
__global__ void dispatch_plan_kernel(const int* __restrict__ topi,
                                     int* __restrict__ dest, int N) {
    int e = threadIdx.x;
    if (e >= NEXP) return;
    int cnt = 0;
    for (int i = 0; i < TOPK * N; i++) {
        int k = i / N, n = i - k * N;
        if (topi[n * 2 + k] == e) {
            dest[i] = (cnt < CAPACITY) ? (e * CAPACITY + cnt) : -1;
            cnt++;
        }
    }
}

// ---------------------------------------------------------------------------
// Scatter token rows (fp32 -> bf16) into dispatch buffer. One block per
// routed entry; dropped entries skip. Unfilled slots are never read back.
// ---------------------------------------------------------------------------
__global__ __launch_bounds__(256)
void scatter_kernel(const float* __restrict__ x, const int* __restrict__ dest,
                    unsigned short* __restrict__ bufA, int N) {
    int i = blockIdx.x;
    int d = dest[i];
    if (d < 0) return;
    int token = i % N;
    const float* src = x + (long)token * HDIM;
    unsigned short* dst = bufA + (long)d * HDIM;
    for (int h = threadIdx.x; h < HDIM; h += 256)
        dst[h] = f32_to_bf16(src[h]);
}

// ---------------------------------------------------------------------------
// bf16 WMMA GEMM:  C[M,N] = A[M,K] @ B[N,K]^T  (+bias, optional GELU)
// 128x128 workgroup tile, 8 waves (2x4), each wave 64x32 = 4x2 WMMA accums.
// Tiles staged via GLOBAL_LOAD_ASYNC_TO_LDS_B128 (ASYNCcnt), fragments via
// ds_load_b128 per the ISA 16-bit A/B VGPR layouts.
// EPI==0: gelu(x+bias) -> bf16 out.  EPI==1: x+bias -> fp32 out.
// ---------------------------------------------------------------------------
__device__ __forceinline__ v16bf load_frag(const unsigned short* tile, int ld) {
    int lane = threadIdx.x & 31;
    int row = lane & 15;
    int kb  = (lane >> 4) << 3;   // lanes 0-15: K 0..7 & 16..23; lanes 16-31: 8..15 & 24..31
    union { v16bf v; uint4 q[2]; } f;
    f.q[0] = *reinterpret_cast<const uint4*>(tile + row * ld + kb);
    f.q[1] = *reinterpret_cast<const uint4*>(tile + row * ld + kb + 16);
    return f.v;
}

template <int EPI>
__global__ __launch_bounds__(256)
void gemm_bf16_kernel(const unsigned short* __restrict__ A,   // [M,K] row-major bf16
                      const unsigned short* __restrict__ B,   // [N,K] row-major bf16 (= B^T)
                      const float* __restrict__ bias,         // [N]
                      void* __restrict__ Cout,
                      int Kdim, int Ndim) {
    constexpr int LD = 40;                                    // padded bf16 stride (80 B rows)
    __shared__ __align__(16) unsigned short As[128 * LD];
    __shared__ __align__(16) unsigned short Bs[128 * LD];

    const int tid  = threadIdx.x;
    const int wave = tid >> 5;
    const int lane = tid & 31;
    const int wm = wave >> 2;            // 0..1  (64-row strip)
    const int wn = wave & 3;             // 0..3  (32-col strip)

    v8f acc[4][2];
#pragma unroll
    for (int i = 0; i < 4; i++)
#pragma unroll
        for (int j = 0; j < 2; j++) {
            v8f z = {0.f, 0.f, 0.f, 0.f, 0.f, 0.f, 0.f, 0.f};
            acc[i][j] = z;
        }

    const int lr = tid >> 1;             // 0..127 tile row
    const int lc = (tid & 1) << 4;       // 0 or 16
    const unsigned short* Ag = A + (long)(blockIdx.y * 128 + lr) * Kdim + lc;
    const unsigned short* Bg = B + (long)(blockIdx.x * 128 + lr) * Kdim + lc;
    unsigned short* Asr = &As[lr * LD + lc];
    unsigned short* Bsr = &Bs[lr * LD + lc];

    for (int k0 = 0; k0 < Kdim; k0 += 32) {
        __syncthreads();                          // previous iteration's reads done
        async_ld_b128(Asr,     Ag + k0);          // 8 bf16 = 16 B per lane, direct to LDS
        async_ld_b128(Asr + 8, Ag + k0 + 8);
        async_ld_b128(Bsr,     Bg + k0);
        async_ld_b128(Bsr + 8, Bg + k0 + 8);
        wait_asynccnt0();                         // this wave's async copies landed
        __syncthreads();                          // all waves' copies landed

        v16bf af[4], bfr[2];
#pragma unroll
        for (int i = 0; i < 4; i++) af[i]  = load_frag(&As[(wm * 64 + i * 16) * LD], LD);
#pragma unroll
        for (int j = 0; j < 2; j++) bfr[j] = load_frag(&Bs[(wn * 32 + j * 16) * LD], LD);
#pragma unroll
        for (int i = 0; i < 4; i++)
#pragma unroll
            for (int j = 0; j < 2; j++)
                acc[i][j] = __builtin_amdgcn_wmma_f32_16x16x32_bf16(
                    false, af[i], false, bfr[j], (short)0, acc[i][j], false, false);
    }

    // Epilogue. C/D layout: VGPR t -> row t (+8 for lanes>=16), col = lane&15.
    const int roff = (lane >> 4) << 3;
    const int col  = lane & 15;
#pragma unroll
    for (int i = 0; i < 4; i++)
#pragma unroll
        for (int j = 0; j < 2; j++) {
            int rbase = blockIdx.y * 128 + wm * 64 + i * 16 + roff;
            int cbase = blockIdx.x * 128 + wn * 32 + j * 16 + col;
            float bv = bias[cbase];
#pragma unroll
            for (int t = 0; t < 8; t++) {
                float v = acc[i][j][t] + bv;
                long idx = (long)(rbase + t) * Ndim + cbase;
                if (EPI == 0) {
                    v = gelu_exact(v);
                    reinterpret_cast<unsigned short*>(Cout)[idx] = f32_to_bf16(v);
                } else {
                    reinterpret_cast<float*>(Cout)[idx] = v;
                }
            }
        }
}

// ---------------------------------------------------------------------------
// Combine: out[n] = sum_k keep * prob * hout[dest].  Deterministic gather.
// ---------------------------------------------------------------------------
__global__ __launch_bounds__(256)
void combine_kernel(const float* __restrict__ hout, const int* __restrict__ dest,
                    const float* __restrict__ topp, float* __restrict__ out, int N) {
    int n = blockIdx.x;
    int d0 = dest[n];
    int d1 = dest[N + n];
    float p0 = topp[n * 2 + 0];
    float p1 = topp[n * 2 + 1];
    for (int h = threadIdx.x; h < HDIM; h += 256) {
        float s = 0.0f;
        if (d0 >= 0) s += p0 * hout[(long)d0 * HDIM + h];
        if (d1 >= 0) s += p1 * hout[(long)d1 * HDIM + h];
        out[(long)n * HDIM + h] = s;
    }
}

// ---------------------------------------------------------------------------
extern "C" void kernel_launch(void* const* d_in, const int* in_sizes, int n_in,
                              void* d_out, int out_size, void* d_ws, size_t ws_size,
                              hipStream_t stream) {
    const float* x   = (const float*)d_in[0];  // [N, H]
    const float* rw  = (const float*)d_in[1];  // [E, H]
    const float* w1  = (const float*)d_in[2];  // [F, H]
    const float* b1  = (const float*)d_in[3];  // [F]
    const float* w2  = (const float*)d_in[4];  // [H, F]
    const float* b2  = (const float*)d_in[5];  // [H]
    float* out = (float*)d_out;

    const int N = in_sizes[0] / HDIM;          // 2048 tokens
    const int SLOTS = NEXP * CAPACITY;         // 4096

    // ---- workspace carve-up (bytes) ----
    char* ws = (char*)d_ws;
    unsigned short* w1bf = (unsigned short*)(ws);                                 // 32 MB
    unsigned short* w2bf = (unsigned short*)(ws + 33554432);                      // 32 MB
    unsigned short* bufA = (unsigned short*)(ws + 67108864);                      // 16 MB
    unsigned short* hmid = (unsigned short*)(ws + 83886080);                      // 64 MB
    float*          hout = (float*)         (ws + 150994944);                     // 32 MB
    int*            topi = (int*)           (ws + 184549376);                     // 16 KB
    float*          topp = (float*)         (ws + 184565760);                     // 16 KB
    int*            dest = (int*)           (ws + 184582144);                     // 16 KB

    // 1) weights fp32 -> bf16 (streamed once; negligible vs GEMM time)
    cvt_bf16_kernel<<<4096, 256, 0, stream>>>(w1, w1bf, (long)FDIM * HDIM / 4);
    cvt_bf16_kernel<<<4096, 256, 0, stream>>>(w2, w2bf, (long)HDIM * FDIM / 4);

    // 2) router: softmax + top-2
    router_kernel<<<(N + 7) / 8, 256, 0, stream>>>(x, rw, topi, topp, N);

    // 3) deterministic capacity scan (slot-major order, matches reference cumsum)
    dispatch_plan_kernel<<<1, 32, 0, stream>>>(topi, dest, N);

    // 4) scatter tokens -> bf16 dispatch buffer
    scatter_kernel<<<TOPK * N, 256, 0, stream>>>(x, dest, bufA, N);

    // 5) GEMM1: hmid = gelu(bufA @ w1^T + b1)   [4096 x 8192], bf16 out
    {
        dim3 grid(FDIM / 128, SLOTS / 128);    // (64, 32)
        gemm_bf16_kernel<0><<<grid, 256, 0, stream>>>(bufA, w1bf, b1, (void*)hmid,
                                                      HDIM, FDIM);
    }
    // 6) GEMM2: hout = hmid @ w2^T + b2        [4096 x 2048], fp32 out
    {
        dim3 grid(HDIM / 128, SLOTS / 128);    // (16, 32)
        gemm_bf16_kernel<1><<<grid, 256, 0, stream>>>(hmid, w2bf, b2, (void*)hout,
                                                      FDIM, HDIM);
    }
    // 7) weighted combine back to token order
    combine_kernel<<<N, 256, 0, stream>>>(hout, dest, topp, out, N);
}
